// GRUNet_57166014709773
// MI455X (gfx1250) — matrix operand order repaired and checked by reading, not compiled, not run on hardware
//
#include <hip/hip_runtime.h>

// GRU: B=64, T=1024, D=64, H=512, O=1. Persistent 8-block scan kernel,
// WMMA bf16 16x16x32, pre-swizzled operand layouts, double-buffered hidden
// state, agent-scope atomic grid barrier per step.
// Round 3: hoist per-wave fragment base pointers out of the K-loops so the
// steady state is load(+immediate)/wmma only — removes v_lshl_add_u64 address
// rebuilds and the WMMA->VALU hazard v_nops they forced.

typedef __attribute__((ext_vector_type(16))) __bf16 v16bf;
typedef __attribute__((ext_vector_type(8)))  float  v8f;

#define Bsz 64
#define Tsz 1024
#define Dsz 64
#define Hsz 512
#define NBLK 8   // 8 blocks x 64 H-cols = 512

__device__ __forceinline__ float sigf(float x){ return 1.0f/(1.0f+__expf(-x)); }

// ---------------- init kernels (run every launch; deterministic) ----------------

// w_hh [1536,512] f32 -> bf16 B-fragment layout: [ntile 0..95][kstep 0..15][lane 0..31][e 0..15]
// B mapping (16-bit 32x16): n = lane&15, k = (e>>3)*16 + (lane>>4)*8 + (e&7)
__global__ void k_swz_whh(const float* __restrict__ w, __bf16* __restrict__ o){
  int id = blockIdx.x*256 + threadIdx.x;           // 786432 total
  int e = id & 15, lane = (id>>4)&31, ks = (id>>9)&15, nt = id>>13;
  int wrow = nt*16 + (lane&15);
  int k = ks*32 + ((e>>3)<<4) + ((lane>>4)<<3) + (e&7);
  o[id] = (__bf16)w[wrow*Hsz + k];
}

// w_ih [1536,64] f32 -> bf16 B-fragment layout: [ntile 0..95][kstep 0..1][lane][e]
__global__ void k_swz_wih(const float* __restrict__ w, __bf16* __restrict__ o){
  int id = blockIdx.x*256 + threadIdx.x;           // 98304 total
  int e = id & 15, lane = (id>>4)&31, ks = (id>>9)&1, nt = id>>10;
  int wrow = nt*16 + (lane&15);
  int k = ks*32 + ((e>>3)<<4) + ((lane>>4)<<3) + (e&7);
  o[id] = (__bf16)w[wrow*Dsz + k];
}

// h0 [64,512] f32 -> f32 master + bf16 A-fragment swizzle into buffer 0; zero barrier ctr.
// A mapping (16-bit 16x32): m = lane&15; k = (e>>3)*16 + (lane>>4)*8 + ((e>>1)&3)*2 + (e&1)
__global__ void k_init_h(const float* __restrict__ h0, float* __restrict__ hf,
                         __bf16* __restrict__ hsw, unsigned* __restrict__ ctr){
  int id = blockIdx.x*256 + threadIdx.x;           // 32768 total
  float v = h0[id];
  hf[id] = v;
  int b = id>>9, k = id&511;
  int mt = b>>4, ks = k>>5, ko = k&31;
  int e  = ((ko>>4)&1)*8 + ((ko>>1)&3)*2 + (ko&1);
  int ln = ((ko>>3)&1)*16 + (b&15);
  hsw[(((mt*16+ks)*32+ln)<<4) + e] = (__bf16)v;
  if (id==0) *ctr = 0u;
}

__global__ void k_init_out(const float* __restrict__ fcb, float* __restrict__ out){
  int id = blockIdx.x*256 + threadIdx.x;           // 65536 total
  out[id] = fcb[0];
}

// ---------------- persistent scan kernel ----------------

__global__ __launch_bounds__(256)
void k_gru_scan(const float* __restrict__ x, const __bf16* __restrict__ Wsw,
                const __bf16* __restrict__ Wxsw, __bf16* __restrict__ hsw,
                float* __restrict__ hf, const float* __restrict__ fcw,
                float* __restrict__ out, unsigned* __restrict__ ctr){
  __shared__ __align__(32) __bf16 xsw[4096];   // x_t A-fragments [mt][s][lane][16]
  __shared__ float Gbuf[32][264];              // planes: r | z | n_h | n_x (64 cols each)
  __shared__ float red[32];                    // fc partial reduction per batch row

  const int tid  = threadIdx.x;
  const int bx   = blockIdx.x;                 // 0..7 -> H cols [bx*64, bx*64+64)
  const int lane = tid & 31;
  const int wv   = tid >> 5;                   // 8 waves
  const int mhalf = wv >> 2;                   // which 32 batch rows
  const int q     = wv & 3;                    // which 16 of the 64 H cols
  const int mt0 = mhalf*2, mt1 = mhalf*2 + 1;
  const int ntR = 0*32 + bx*4 + q;             // global 16-row tiles of w (r/z/n gates)
  const int ntZ = 1*32 + bx*4 + q;
  const int ntN = 2*32 + bx*4 + q;

  // ---- hoisted per-wave fragment base pointers (loop-invariant) ----
  // fragment address = nt*8192 + s*512 + lane*16  (elements)
  const __bf16* bRb = Wsw  + ntR*8192 + lane*16;
  const __bf16* bZb = Wsw  + ntZ*8192 + lane*16;
  const __bf16* bNb = Wsw  + ntN*8192 + lane*16;
  const __bf16* xRb = Wxsw + ntR*1024 + lane*16;   // x-proj: nt*1024 + s*512 + lane*16
  const __bf16* xZb = Wxsw + ntZ*1024 + lane*16;
  const __bf16* xNb = Wxsw + ntN*1024 + lane*16;
  const __bf16* xa0 = &xsw[mt0*1024 + lane*16];
  const __bf16* xa1 = &xsw[mt1*1024 + lane*16];
  const int aoff0 = mt0*8192 + lane*16;            // into hR: + s*512
  const int aoff1 = mt1*8192 + lane*16;

  // x staging: thread tid fills lane-fragment tid ( [mt=tid>>6][s=(tid>>5)&1][lane=tid&31] )
  const int f_b = ((tid>>6)<<4) + (tid&15);
  const float* xrow = x + (size_t)f_b * Tsz * Dsz + (((tid>>5)&1)<<5) + (((tid>>4)&1)<<3);

  const v8f zero8 = {0.f,0.f,0.f,0.f,0.f,0.f,0.f,0.f};

  for (int t=0; t<Tsz; ++t){
    // ---- stage x_t into A-fragment layout (bf16) in LDS ----
    {
      const float* p = xrow + (size_t)t*Dsz;
      float4 r0 = *(const float4*)(p);       // k = khalf*8 + 0..3
      float4 r1 = *(const float4*)(p+4);     // k = khalf*8 + 4..7
      float4 r2 = *(const float4*)(p+16);    // k = 16 + khalf*8 + 0..3
      float4 r3 = *(const float4*)(p+20);
      __bf16* d = &xsw[tid<<4];
      d[0]=(__bf16)r0.x; d[1]=(__bf16)r0.y; d[2]=(__bf16)r0.z; d[3]=(__bf16)r0.w;
      d[4]=(__bf16)r1.x; d[5]=(__bf16)r1.y; d[6]=(__bf16)r1.z; d[7]=(__bf16)r1.w;
      d[8]=(__bf16)r2.x; d[9]=(__bf16)r2.y; d[10]=(__bf16)r2.z; d[11]=(__bf16)r2.w;
      d[12]=(__bf16)r3.x; d[13]=(__bf16)r3.y; d[14]=(__bf16)r3.z; d[15]=(__bf16)r3.w;
    }
    __syncthreads();

    const __bf16* hR = hsw + ((t&1)<<15);       // read state t
    __bf16*       hW = hsw + (((t+1)&1)<<15);   // write state t+1
    const __bf16* aB0 = hR + aoff0;
    const __bf16* aB1 = hR + aoff1;

    v8f ar0=zero8, ar1=zero8, az0=zero8, az1=zero8;
    v8f anh0=zero8, anh1=zero8, anx0=zero8, anx1=zero8;

    // ---- recurrence GEMM: [64x512] x [512x192], K in 16 steps of 32 ----
    // unroll capped at 2: one-iteration load-ahead pipeline without spilling.
    #pragma unroll 2
    for (int s=0; s<16; ++s){
      v16bf a0 = *(const v16bf*)(aB0 + s*512);
      v16bf a1 = *(const v16bf*)(aB1 + s*512);
      v16bf b0 = *(const v16bf*)(bRb + s*512);
      v16bf b1 = *(const v16bf*)(bZb + s*512);
      v16bf b2 = *(const v16bf*)(bNb + s*512);
      ar0  = __builtin_amdgcn_wmma_f32_16x16x32_bf16(false,a0,false,b0,(short)0,ar0 ,false,false);
      ar1  = __builtin_amdgcn_wmma_f32_16x16x32_bf16(false,a1,false,b0,(short)0,ar1 ,false,false);
      az0  = __builtin_amdgcn_wmma_f32_16x16x32_bf16(false,a0,false,b1,(short)0,az0 ,false,false);
      az1  = __builtin_amdgcn_wmma_f32_16x16x32_bf16(false,a1,false,b1,(short)0,az1 ,false,false);
      anh0 = __builtin_amdgcn_wmma_f32_16x16x32_bf16(false,a0,false,b2,(short)0,anh0,false,false);
      anh1 = __builtin_amdgcn_wmma_f32_16x16x32_bf16(false,a1,false,b2,(short)0,anh1,false,false);
    }
    // ---- input projection GEMM: [64x64] x [64x192], K in 2 steps ----
    #pragma unroll 1
    for (int s=0; s<2; ++s){
      v16bf a0 = *(const v16bf*)(xa0 + s*512);
      v16bf a1 = *(const v16bf*)(xa1 + s*512);
      v16bf b0 = *(const v16bf*)(xRb + s*512);
      v16bf b1 = *(const v16bf*)(xZb + s*512);
      v16bf b2 = *(const v16bf*)(xNb + s*512);
      ar0  = __builtin_amdgcn_wmma_f32_16x16x32_bf16(false,a0,false,b0,(short)0,ar0 ,false,false);
      ar1  = __builtin_amdgcn_wmma_f32_16x16x32_bf16(false,a1,false,b0,(short)0,ar1 ,false,false);
      az0  = __builtin_amdgcn_wmma_f32_16x16x32_bf16(false,a0,false,b1,(short)0,az0 ,false,false);
      az1  = __builtin_amdgcn_wmma_f32_16x16x32_bf16(false,a1,false,b1,(short)0,az1 ,false,false);
      anx0 = __builtin_amdgcn_wmma_f32_16x16x32_bf16(false,a0,false,b2,(short)0,anx0,false,false);
      anx1 = __builtin_amdgcn_wmma_f32_16x16x32_bf16(false,a1,false,b2,(short)0,anx1,false,false);
    }

    // ---- gate combine in two 32-row passes through LDS ----
    for (int hm=0; hm<2; ++hm){
      __syncthreads();
      if (mhalf == hm){
        int cb = (q<<4) + (lane&15);
        int r0 = ((lane>>4)&1)<<3;
        #pragma unroll
        for (int i=0;i<8;++i){
          Gbuf[r0+i][cb]         = ar0[i];
          Gbuf[r0+i][64+cb]      = az0[i];
          Gbuf[r0+i][128+cb]     = anh0[i];
          Gbuf[r0+i][192+cb]     = anx0[i];
          Gbuf[16+r0+i][cb]      = ar1[i];
          Gbuf[16+r0+i][64+cb]   = az1[i];
          Gbuf[16+r0+i][128+cb]  = anh1[i];
          Gbuf[16+r0+i][192+cb]  = anx1[i];
        }
      }
      if (tid < 32) red[tid] = 0.f;
      __syncthreads();
      {
        int bl = tid & 31, jg = tid >> 5;
        int b  = hm*32 + bl;
        float p = 0.f;
        #pragma unroll
        for (int jj=0; jj<8; ++jj){
          int jl = (jg<<3) + jj;
          float rv = sigf(Gbuf[bl][jl]);
          float zv = sigf(Gbuf[bl][64+jl]);
          float nv = tanhf(Gbuf[bl][192+jl] + rv*Gbuf[bl][128+jl]);
          int col = (bx<<6) + jl;
          float ho = hf[b*Hsz + col];
          float hn = (1.f - zv)*nv + zv*ho;
          hf[b*Hsz + col] = hn;
          int ks = col>>5, ko = col&31;
          int e  = ((ko>>4)&1)*8 + ((ko>>1)&3)*2 + (ko&1);
          int ln = ((ko>>3)&1)*16 + (b&15);
          hW[((((b>>4)*16+ks)*32)+ln)*16 + e] = (__bf16)hn;
          p += fmaxf(hn, 0.f) * fcw[col];
        }
        atomicAdd(&red[bl], p);
      }
      __syncthreads();
      if (tid < 32) atomicAdd(&out[(size_t)(hm*32+tid)*Tsz + t], red[tid]);
    }
    __syncthreads();

    // prefetch next timestep's x rows (global_prefetch)
    if (t+1 < Tsz){
      const float* pf = x + ((size_t)(tid>>2)*Tsz + (t+1))*Dsz + ((tid&3)<<4);
      __builtin_prefetch(pf, 0, 1);
    }

    // ---- inter-block step barrier ----
    __threadfence();
    __syncthreads();
    __builtin_amdgcn_s_cluster_barrier();   // architecturally NOP when ClusterID==0
    if (tid == 0){
      __hip_atomic_fetch_add(ctr, 1u, __ATOMIC_RELEASE, __HIP_MEMORY_SCOPE_AGENT);
      unsigned tgt = (unsigned)(NBLK*(t+1));
      while (__hip_atomic_load(ctr, __ATOMIC_ACQUIRE, __HIP_MEMORY_SCOPE_AGENT) < tgt)
        __builtin_amdgcn_s_sleep(1);
    }
    __syncthreads();
  }

  // ---- final hidden state: d_out[B*T : B*T + B*H] = h_T ----
  for (int i = tid; i < 4096; i += 256){
    int b = i>>6, jl = i&63, col = (bx<<6) + jl;
    out[(size_t)Bsz*Tsz + (size_t)b*Hsz + col] = hf[b*Hsz + col];
  }
}

// ---------------- host launcher ----------------

extern "C" void kernel_launch(void* const* d_in, const int* in_sizes, int n_in,
                              void* d_out, int out_size, void* d_ws, size_t ws_size,
                              hipStream_t stream) {
  (void)in_sizes; (void)n_in; (void)out_size; (void)ws_size;
  const float* x    = (const float*)d_in[0];
  const float* h0   = (const float*)d_in[1];
  const float* w_ih = (const float*)d_in[2];
  const float* w_hh = (const float*)d_in[3];
  const float* fc_w = (const float*)d_in[4];
  const float* fc_b = (const float*)d_in[5];
  float* out = (float*)d_out;

  char* ws = (char*)d_ws;
  __bf16* Wsw  = (__bf16*)(ws);                 // 786432 bf16 = 1,572,864 B
  __bf16* Wxsw = (__bf16*)(ws + 1572864);       //  98304 bf16 =   196,608 B
  __bf16* hsw  = (__bf16*)(ws + 1769472);       // 2 x 32768 bf16 = 131,072 B
  float*  hf   = (float*)(ws + 1900544);        // 32768 f32 = 131,072 B
  unsigned* ctr = (unsigned*)(ws + 2031616);

  k_swz_whh <<<3072, 256, 0, stream>>>(w_hh, Wsw);
  k_swz_wih <<< 384, 256, 0, stream>>>(w_ih, Wxsw);
  k_init_h  <<< 128, 256, 0, stream>>>(h0, hf, hsw, ctr);
  k_init_out<<< 256, 256, 0, stream>>>(fc_b, out);
  k_gru_scan<<<NBLK, 256, 0, stream>>>(x, Wsw, Wxsw, hsw, hf, fc_w, out, ctr);
}
